// PDANet_7395933684150
// MI455X (gfx1250) — matrix operand order repaired and checked by reading, not compiled, not run on hardware
//
#include <hip/hip_runtime.h>
#include <hip/hip_bf16.h>

// ---------------------------------------------------------------------------
// PDANet on MI455X (gfx1250): all GEMM-like work via v_wmma_f32_16x16x32_f16.
// fp16 activations/weights, fp32 accumulate + bias/softmax statistics.
// This round: 2x4 register blocking per wave (32 pixels x 64 cols): each
// A-frag feeds 4 WMMAs, each B-frag 2 WMMAs; 8 WMMAs per K-step, 16 per
// unrolled iteration. Conv adds global_prefetch_b8 of the next weight slice.
// ---------------------------------------------------------------------------

typedef _Float16 h16;
typedef __attribute__((ext_vector_type(16))) _Float16 v16h;
typedef __attribute__((ext_vector_type(8)))  _Float16 v8h;
typedef __attribute__((ext_vector_type(8)))  float    v8f;

#define CONV_NONE  0
#define CONV_RELU  1
#define CONV_PRELU 2

__device__ static inline v16h zero_v16h() {
  v16h z;
#pragma unroll
  for (int i = 0; i < 16; ++i) z[i] = (_Float16)0;
  return z;
}

__device__ static inline v16h pack_v16h(v8h lo, v8h hi) {
  v16h r;
#pragma unroll
  for (int i = 0; i < 8; ++i) { r[i] = lo[i]; r[i + 8] = hi[i]; }
  return r;
}

// A-fragment (16x32 f16, row-major source, ld in elements, K%32 step).
// ISA layout: lanes 0-15 row=lane, elems = K[0..7],K[16..23];
//             lanes 16-31 row=lane-16, elems = K[8..15],K[24..31].
__device__ static inline v16h load_frag_a(const h16* __restrict__ base, int ld,
                                          int row, bool valid, int k0) {
  if (!valid) return zero_v16h();
  const int half = (threadIdx.x & 31) >> 4;
  const h16* p = base + (size_t)row * ld + k0 + half * 8;
  v8h lo = *(const v8h*)p;
  v8h hi = *(const v8h*)(p + 16);
  return pack_v16h(lo, hi);
}

// B-fragment (32x16 f16). Source stored [N][K] row-major (K contiguous).
// ISA layout: col = lane&15; lanes 0-15 hold K[0..7],K[16..23], lanes 16-31 K[8..15],K[24..31].
__device__ static inline v16h load_frag_b(const h16* __restrict__ base, int ld,
                                          int col, bool valid, int k0) {
  if (!valid) return zero_v16h();
  const int half = (threadIdx.x & 31) >> 4;
  const h16* p = base + (size_t)col * ld + k0 + half * 8;
  v8h lo = *(const v8h*)p;
  v8h hi = *(const v8h*)(p + 16);
  return pack_v16h(lo, hi);
}

__device__ static inline v8f wmma_f16(v16h a, v16h b, v8f c) {
  return __builtin_amdgcn_wmma_f32_16x16x32_f16(false, a, false, b, (short)0, c,
                                                false, false);
}

// ---------------------------------------------------------------------------
// Implicit-GEMM conv (ksize 1 or 3, pad = ksize/2), fp16 in/out, fused epilogue.
// Block: 64 threads = 2 waves. Wave tile = 32 pixels x 64 output channels.
// Block tile = 32 pixels x 128 channels. LDS im2col tile = 32 x 64 (K),
// rows padded to 72 halves (lane l starts at bank group 9*l mod 16 -> no
// conflicts on ds_read_b128). Weights fp16 [Cout][Kpad], Kpad % 64 == 0.
// ---------------------------------------------------------------------------
__global__ void conv_wmma_k(const h16* __restrict__ in, const h16* __restrict__ w,
                            const float* __restrict__ bias,
                            const float* __restrict__ alpha,
                            const h16* __restrict__ resid,
                            h16* __restrict__ out16, float* __restrict__ out32,
                            int Cin, int Cout, int H, int W, int ksize, int Kpad,
                            int mode) {
  __shared__ __align__(16) h16 As[32][72];
  const int HW = H * W;
  const int mBase = blockIdx.x * 32;
  const int wave = threadIdx.x >> 5;
  const int lane = threadIdx.x & 31;
  const int halfid = lane >> 4;
  const int r = lane & 15;
  const int colBase = blockIdx.y * 128 + wave * 64;
  const int K = (ksize == 3) ? Cin * 9 : Cin;

  v8f acc[2][4];
#pragma unroll
  for (int e = 0; e < 2; ++e)
#pragma unroll
    for (int f = 0; f < 4; ++f)
#pragma unroll
      for (int i = 0; i < 8; ++i) acc[e][f][i] = 0.f;

  for (int k0 = 0; k0 < Kpad; k0 += 64) {
    // cooperative im2col of 32 pixels x 64 K into LDS
    for (int f = threadIdx.x; f < 2048; f += blockDim.x) {
      const int row = f >> 6, kk = f & 63;
      const int k = k0 + kk;
      const int p = mBase + row;
      h16 v = (h16)0;
      if (p < HW && k < K) {
        const int y = p / W, x = p - y * W;
        if (ksize == 3) {
          const int ci = k / 9;
          const int rr = k - ci * 9;
          const int iy = y + rr / 3 - 1;
          const int ix = x + (rr % 3) - 1;
          if (iy >= 0 && iy < H && ix >= 0 && ix < W)
            v = in[(size_t)ci * HW + (size_t)iy * W + ix];
        } else {
          v = in[(size_t)k * HW + p];
        }
      }
      As[row][kk] = v;
    }
    // prefetch next weight K-slice while this tile is consumed
    if (k0 + 64 < Kpad) {
      const int pc = colBase + lane;
      if (pc < Cout)
        __builtin_prefetch(w + (size_t)pc * Kpad + k0 + 64, 0, 1);
    }
    __syncthreads();
#pragma unroll
    for (int kk = 0; kk < 64; kk += 32) {
      v16h a0 = load_frag_a(&As[0][0], 72, r, true, kk);
      v16h a1 = load_frag_a(&As[0][0], 72, r + 16, true, kk);
#pragma unroll
      for (int f = 0; f < 4; ++f) {
        const int col = colBase + f * 16 + r;
        v16h b = load_frag_b(w, Kpad, col, col < Cout, k0 + kk);
        acc[0][f] = wmma_f16(a0, b, acc[0][f]);
        acc[1][f] = wmma_f16(a1, b, acc[1][f]);
      }
    }
    __syncthreads();
  }

  const float av = (mode == CONV_PRELU && alpha) ? alpha[0] : 0.f;
#pragma unroll
  for (int e = 0; e < 2; ++e) {
#pragma unroll
    for (int f = 0; f < 4; ++f) {
#pragma unroll
      for (int j = 0; j < 8; ++j) {
        const int p = mBase + e * 16 + j + halfid * 8;
        const int c = colBase + f * 16 + r;
        if (p < HW && c < Cout) {
          float v = acc[e][f][j] + (bias ? bias[c] : 0.f);
          if (mode == CONV_RELU)       v = v > 0.f ? v : 0.f;
          else if (mode == CONV_PRELU) v = v >= 0.f ? v : av * v;
          if (resid) v += (float)resid[(size_t)c * HW + p];
          if (out16) out16[(size_t)c * HW + p] = (h16)v;
          if (out32) out32[(size_t)c * HW + p] = v;
        }
      }
    }
  }
}

// ---------------------------------------------------------------------------
// D[M][ldd] = A[M][K] * B[N][K]^T  (scores GEMM, K % 32 == 0, fp16 out)
// Block: 128 threads = 4 waves; wave tile = 32 x 64; block tile = 32 x 256.
// ---------------------------------------------------------------------------
__global__ void gemm_nt_wmma_k(const h16* __restrict__ A, const h16* __restrict__ B,
                               h16* __restrict__ D, int M, int N, int K, int ldd) {
  const int mBase = blockIdx.x * 32;
  const int wave = threadIdx.x >> 5;
  const int lane = threadIdx.x & 31;
  const int halfid = lane >> 4;
  const int r = lane & 15;
  const int nBase = blockIdx.y * 256 + wave * 64;
  const int row0 = mBase + r;
  const int row1 = mBase + 16 + r;
  const bool r0v = row0 < M;
  const bool r1v = row1 < M;

  v8f acc[2][4];
#pragma unroll
  for (int e = 0; e < 2; ++e)
#pragma unroll
    for (int f = 0; f < 4; ++f)
#pragma unroll
      for (int i = 0; i < 8; ++i) acc[e][f][i] = 0.f;

#pragma unroll 2
  for (int k0 = 0; k0 < K; k0 += 32) {
    v16h a0 = load_frag_a(A, K, row0, r0v, k0);
    v16h a1 = load_frag_a(A, K, row1, r1v, k0);
#pragma unroll
    for (int f = 0; f < 4; ++f) {
      const int col = nBase + f * 16 + r;
      v16h b = load_frag_b(B, K, col, col < N, k0);
      acc[0][f] = wmma_f16(a0, b, acc[0][f]);
      acc[1][f] = wmma_f16(a1, b, acc[1][f]);
    }
  }
#pragma unroll
  for (int e = 0; e < 2; ++e) {
#pragma unroll
    for (int f = 0; f < 4; ++f) {
#pragma unroll
      for (int j = 0; j < 8; ++j) {
        const int p = mBase + e * 16 + j + halfid * 8;
        const int n = nBase + f * 16 + r;
        if (p < M && n < N) D[(size_t)p * ldd + n] = (h16)acc[e][f][j];
      }
    }
  }
}

// ---------------------------------------------------------------------------
// Pyramid attention aggregation: transposed 3x3 conv of attn with R, as 9
// spatially-shifted GEMMs. attn: [HW][Lpad] fp16, R: [9][C][Lpad] fp16.
// out[c][p] = 0.25 * sum_t sum_l attn[shift_t(p)][l] * R[t][c][l] + resid[c][p]
// Block: 64 threads = 2 waves; wave tile = 32 pixels x 64 channels.
// ---------------------------------------------------------------------------
__global__ void pa_aggregate_k(const h16* __restrict__ attn, const h16* __restrict__ R,
                               const h16* __restrict__ resid, h16* __restrict__ out,
                               int C, int H, int W, int Lpad) {
  const int HW = H * W;
  const int mBase = blockIdx.x * 32;
  const int wave = threadIdx.x >> 5;
  const int lane = threadIdx.x & 31;
  const int halfid = lane >> 4;
  const int r = lane & 15;
  const int cBase = blockIdx.y * 128 + wave * 64;
  const int p0 = mBase + r;
  const int p1 = mBase + 16 + r;
  const int y0 = (p0 < HW) ? p0 / W : 0;
  const int x0 = (p0 < HW) ? p0 - y0 * W : 0;
  const int y1 = (p1 < HW) ? p1 / W : 0;
  const int x1 = (p1 < HW) ? p1 - y1 * W : 0;

  v8f acc[2][4];
#pragma unroll
  for (int e = 0; e < 2; ++e)
#pragma unroll
    for (int f = 0; f < 4; ++f)
#pragma unroll
      for (int i = 0; i < 8; ++i) acc[e][f][i] = 0.f;

  for (int t = 0; t < 9; ++t) {
    const int dy = t / 3 - 1, dx = t % 3 - 1;
    const int qy0 = y0 + dy, qx0 = x0 + dx;
    const int qy1 = y1 + dy, qx1 = x1 + dx;
    const bool q0v = (p0 < HW) && qy0 >= 0 && qy0 < H && qx0 >= 0 && qx0 < W;
    const bool q1v = (p1 < HW) && qy1 >= 0 && qy1 < H && qx1 >= 0 && qx1 < W;
    const int q0 = qy0 * W + qx0;
    const int q1 = qy1 * W + qx1;
    const h16* Rt = R + (size_t)t * C * Lpad;
#pragma unroll 2
    for (int l0 = 0; l0 < Lpad; l0 += 32) {
      v16h a0 = load_frag_a(attn, Lpad, q0, q0v, l0);
      v16h a1 = load_frag_a(attn, Lpad, q1, q1v, l0);
#pragma unroll
      for (int f = 0; f < 4; ++f) {
        const int c = cBase + f * 16 + r;
        v16h b = load_frag_b(Rt, Lpad, c, c < C, l0);
        acc[0][f] = wmma_f16(a0, b, acc[0][f]);
        acc[1][f] = wmma_f16(a1, b, acc[1][f]);
      }
    }
  }
#pragma unroll
  for (int e = 0; e < 2; ++e) {
#pragma unroll
    for (int f = 0; f < 4; ++f) {
#pragma unroll
      for (int j = 0; j < 8; ++j) {
        const int pp = mBase + e * 16 + j + halfid * 8;
        const int cc = cBase + f * 16 + r;
        if (pp < HW && cc < C) {
          float v = acc[e][f][j] * 0.25f + (float)resid[(size_t)cc * HW + pp];
          out[(size_t)cc * HW + pp] = (h16)v;
        }
      }
    }
  }
}

// ---------------------------------------------------------------------------
// Softmax (two-pass, scale = 10), over L per pixel row of [HW][Lpad] scores.
// ---------------------------------------------------------------------------
__global__ void softmax_ms_k(const h16* __restrict__ sc, float* __restrict__ m,
                             float* __restrict__ s, int HW, int L, int Lpad) {
  const int p = blockIdx.x * blockDim.x + threadIdx.x;
  if (p >= HW) return;
  const h16* row = sc + (size_t)p * Lpad;
  float mx = -1e30f;
  for (int l = 0; l < L; ++l) mx = fmaxf(mx, 10.f * (float)row[l]);
  float ss = 0.f;
  for (int l = 0; l < L; ++l) ss += __expf(10.f * (float)row[l] - mx);
  m[p] = mx;
  s[p] = ss;
}

__global__ void softmax_apply_k(h16* __restrict__ sc, const float* __restrict__ m,
                                const float* __restrict__ s, int HW, int L, int Lpad) {
  const size_t i = (size_t)blockIdx.x * blockDim.x + threadIdx.x;
  const size_t tot = (size_t)HW * Lpad;
  if (i >= tot) return;
  const int p = (int)(i / Lpad);
  const int l = (int)(i % Lpad);
  float v = 0.f;
  if (l < L) v = __expf(10.f * (float)sc[i] - m[p]) / s[p];
  sc[i] = (h16)v;
}

// ---------------------------------------------------------------------------
// Attention operand builders
// ---------------------------------------------------------------------------
// Q[p][k] = im2col 3x3 of match_base (Chalf channels), Kw = Chalf*9 (mult 64)
__global__ void build_q_k(const h16* __restrict__ mb, h16* __restrict__ Q,
                          int Chalf, int H, int W, int Kw) {
  const size_t i = (size_t)blockIdx.x * blockDim.x + threadIdx.x;
  const size_t tot = (size_t)H * W * Kw;
  if (i >= tot) return;
  const int k = (int)(i % Kw);
  const int p = (int)(i / Kw);
  const int y = p / W, x = p - y * W;
  const int ci = k / 9, rr = k - ci * 9;
  const int iy = y + rr / 3 - 1, ix = x + (rr % 3) - 1;
  h16 v = (h16)0;
  if (iy >= 0 && iy < H && ix >= 0 && ix < W)
    v = mb[(size_t)ci * H * W + (size_t)iy * W + ix];
  Q[i] = v;
}

// Wn row l (= loff + ys*Ws+xs): normalized 3x3 patch of refm at scale s.
// One block (64 threads) per l: reduce sum-of-squares, then write wi/norm.
__global__ void build_wn_k(const h16* __restrict__ refm, h16* __restrict__ Wn,
                           int Hs, int Wsz, int Kw, int loff) {
  const int lidx = blockIdx.x;
  const int ys = lidx / Wsz, xs = lidx - ys * Wsz;
  __shared__ float red[64];
  float ss = 0.f;
  for (int k = threadIdx.x; k < Kw; k += 64) {
    const int ci = k / 9, rr = k - ci * 9;
    const int iy = ys + rr / 3 - 1, ix = xs + (rr % 3) - 1;
    float v = 0.f;
    if (iy >= 0 && iy < Hs && ix >= 0 && ix < Wsz)
      v = (float)refm[(size_t)ci * Hs * Wsz + (size_t)iy * Wsz + ix];
    ss += v * v;
  }
  red[threadIdx.x] = ss;
  __syncthreads();
  for (int st = 32; st > 0; st >>= 1) {
    if (threadIdx.x < st) red[threadIdx.x] += red[threadIdx.x + st];
    __syncthreads();
  }
  const float inv = 1.f / fmaxf(sqrtf(red[0]), 1e-4f);
  h16* outrow = Wn + (size_t)(loff + lidx) * Kw;
  for (int k = threadIdx.x; k < Kw; k += 64) {
    const int ci = k / 9, rr = k - ci * 9;
    const int iy = ys + rr / 3 - 1, ix = xs + (rr % 3) - 1;
    float v = 0.f;
    if (iy >= 0 && iy < Hs && ix >= 0 && ix < Wsz)
      v = (float)refm[(size_t)ci * Hs * Wsz + (size_t)iy * Wsz + ix];
    outrow[k] = (h16)(v * inv);
  }
}

// R[t][c][loff + ls] = base[c][ys + 1 - t/3][xs + 1 - t%3]  (flipped kernel)
__global__ void build_r_k(const h16* __restrict__ base, h16* __restrict__ R,
                          int C, int Hs, int Wsz, int Lpad, int loff) {
  const size_t i = (size_t)blockIdx.x * blockDim.x + threadIdx.x;
  const size_t tot = (size_t)9 * C * Hs * Wsz;
  if (i >= tot) return;
  const int ls = (int)(i % ((size_t)Hs * Wsz));
  const int rem = (int)(i / ((size_t)Hs * Wsz));
  const int c = rem % C;
  const int t = rem / C;
  const int ys = ls / Wsz, xs = ls - ys * Wsz;
  const int iy = ys + 1 - t / 3, ix = xs + 1 - t % 3;
  h16 v = (h16)0;
  if (iy >= 0 && iy < Hs && ix >= 0 && ix < Wsz)
    v = base[(size_t)c * Hs * Wsz + (size_t)iy * Wsz + ix];
  R[(size_t)(t * C + c) * Lpad + loff + ls] = v;
}

// ---------------------------------------------------------------------------
// Elementwise helpers
// ---------------------------------------------------------------------------
__global__ void cast16_k(const float* __restrict__ in, h16* __restrict__ out, size_t n) {
  const size_t i = (size_t)blockIdx.x * blockDim.x + threadIdx.x;
  if (i < n) out[i] = (h16)in[i];
}

__global__ void pack_w_k(const float* __restrict__ w, h16* __restrict__ o,
                         int Cout, int K, int Kpad) {
  const size_t i = (size_t)blockIdx.x * blockDim.x + threadIdx.x;
  const size_t tot = (size_t)Cout * Kpad;
  if (i >= tot) return;
  const int k = (int)(i % Kpad);
  const int c = (int)(i / Kpad);
  o[i] = (k < K) ? (h16)w[(size_t)c * K + k] : (h16)0;
}

__global__ void fill0_k(h16* __restrict__ p, size_t n) {
  const size_t i = (size_t)blockIdx.x * blockDim.x + threadIdx.x;
  if (i < n) p[i] = (h16)0;
}

__global__ void copy16_k(const h16* __restrict__ in, h16* __restrict__ out, size_t n) {
  const size_t i = (size_t)blockIdx.x * blockDim.x + threadIdx.x;
  if (i < n) out[i] = in[i];
}

__global__ void maxpool2_k(const h16* __restrict__ in, h16* __restrict__ out,
                           int C, int H, int W) {
  const int OH = H / 2, OW = W / 2;
  const size_t i = (size_t)blockIdx.x * blockDim.x + threadIdx.x;
  const size_t tot = (size_t)C * OH * OW;
  if (i >= tot) return;
  const int ox = (int)(i % OW);
  const int t = (int)(i / OW);
  const int oy = t % OH;
  const int c = t / OH;
  const h16* ch = in + (size_t)c * H * W;
  const float a0 = (float)ch[(size_t)(2 * oy) * W + 2 * ox];
  const float a1 = (float)ch[(size_t)(2 * oy) * W + 2 * ox + 1];
  const float a2 = (float)ch[(size_t)(2 * oy + 1) * W + 2 * ox];
  const float a3 = (float)ch[(size_t)(2 * oy + 1) * W + 2 * ox + 1];
  out[i] = (h16)fmaxf(fmaxf(a0, a1), fmaxf(a2, a3));
}

__global__ void bilinear_k(const h16* __restrict__ in, h16* __restrict__ out,
                           int C, int H, int W, int OH, int OW, int align) {
  const size_t i = (size_t)blockIdx.x * blockDim.x + threadIdx.x;
  const size_t tot = (size_t)C * OH * OW;
  if (i >= tot) return;
  const int ox = (int)(i % OW);
  const int t = (int)(i / OW);
  const int oy = t % OH;
  const int c = t / OH;
  float ys, xs;
  if (align) {
    ys = (OH > 1) ? (float)oy * (float)(H - 1) / (float)(OH - 1) : 0.f;
    xs = (OW > 1) ? (float)ox * (float)(W - 1) / (float)(OW - 1) : 0.f;
  } else {
    ys = fminf(fmaxf(((float)oy + 0.5f) * (float)H / (float)OH - 0.5f, 0.f), (float)(H - 1));
    xs = fminf(fmaxf(((float)ox + 0.5f) * (float)W / (float)OW - 0.5f, 0.f), (float)(W - 1));
  }
  const int y0 = (int)floorf(ys);
  const int y1 = min(y0 + 1, H - 1);
  const float wy = ys - (float)y0;
  const int x0 = (int)floorf(xs);
  const int x1 = min(x0 + 1, W - 1);
  const float wx = xs - (float)x0;
  const h16* ch = in + (size_t)c * H * W;
  const float v =
      (1.f - wy) * ((1.f - wx) * (float)ch[(size_t)y0 * W + x0] + wx * (float)ch[(size_t)y0 * W + x1]) +
      wy * ((1.f - wx) * (float)ch[(size_t)y1 * W + x0] + wx * (float)ch[(size_t)y1 * W + x1]);
  out[i] = (h16)v;
}

// ---------------------------------------------------------------------------
// Host-side orchestration
// ---------------------------------------------------------------------------
static inline int cdiv(int a, int b) { return (a + b - 1) / b; }
static inline size_t cdivz(size_t a, size_t b) { return (a + b - 1) / b; }

struct Arena {
  char* base;
  size_t off;
  size_t cap;
};
static void* aalloc(Arena& a, size_t bytes) {
  size_t al = (a.off + 255) & ~(size_t)255;
  a.off = al + bytes;
  return a.base + al;
}

static h16* pack_weight(hipStream_t st, Arena& ws, const float* w, int Cout, int K) {
  const int Kpad = ((K + 63) / 64) * 64;
  h16* o = (h16*)aalloc(ws, (size_t)Cout * Kpad * sizeof(h16));
  const size_t n = (size_t)Cout * Kpad;
  pack_w_k<<<(unsigned)cdivz(n, 256), 256, 0, st>>>(w, o, Cout, K, Kpad);
  return o;
}

static void conv(hipStream_t st, const h16* in, const h16* w16, const float* bias,
                 const float* alpha, const h16* resid, h16* out16, float* out32,
                 int Cin, int Cout, int H, int W, int ks, int mode) {
  const int K = (ks == 3) ? Cin * 9 : Cin;
  const int Kpad = ((K + 63) / 64) * 64;
  dim3 g(cdiv(H * W, 32), cdiv(Cout, 128));
  conv_wmma_k<<<g, 64, 0, st>>>(in, w16, bias, alpha, resid, out16, out32, Cin,
                                Cout, H, W, ks, Kpad, mode);
}

struct DownW {
  const float *bb_b, *as_a, *as_b, *m_a, *m_b, *mb_a, *mb_b;
  const float *r1c1_b, *r1c2_b, *r2c1_b, *r2c2_b;
  h16 *bb_w, *as_w, *m_w, *mb_w, *r1c1_w, *r1c2_w, *r2c1_w, *r2c2_w;
};

static DownW pack_down(hipStream_t st, Arena& ws, void* const* din, int b, int Cin,
                       int Cout) {
  auto F = [&](int i) { return (const float*)din[i]; };
  DownW d;
  d.bb_b = F(b + 0);  d.bb_w = pack_weight(st, ws, F(b + 1), Cout, Cin * 9);
  d.as_a = F(b + 2);  d.as_b = F(b + 3);  d.as_w = pack_weight(st, ws, F(b + 4), Cout, Cout);
  d.m_a = F(b + 5);   d.m_b = F(b + 6);   d.m_w = pack_weight(st, ws, F(b + 7), Cout / 2, Cout);
  d.mb_a = F(b + 8);  d.mb_b = F(b + 9);  d.mb_w = pack_weight(st, ws, F(b + 10), Cout / 2, Cout);
  d.r1c1_b = F(b + 11); d.r1c1_w = pack_weight(st, ws, F(b + 12), Cout, Cout * 9);
  d.r1c2_b = F(b + 13); d.r1c2_w = pack_weight(st, ws, F(b + 14), Cout, Cout * 9);
  d.r2c1_b = F(b + 15); d.r2c1_w = pack_weight(st, ws, F(b + 16), Cout, Cout * 9);
  d.r2c2_b = F(b + 17); d.r2c2_w = pack_weight(st, ws, F(b + 18), Cout, Cout * 9);
  return d;
}

// Pyramid attention on x [C,H,W] -> out [C,H,W]
static void pyramid_attn(hipStream_t st, Arena& ar, const h16* x, h16* out, int C,
                         int H, int W, int level, const DownW& d) {
  const int HW = H * W;
  const int Ch = C / 2;
  const int Kw = Ch * 9;  // multiple of 64 for all stages (Ch in {64,128,192})
  int Hs[4], Wv[4], loff[4], L = 0;
  for (int i = 0; i < level; ++i) {
    Hs[i] = (H * (10 - i)) / 10;
    Wv[i] = (W * (10 - i)) / 10;
    loff[i] = L;
    L += Hs[i] * Wv[i];
  }
  const int Lpad = ((L + 31) / 32) * 32;

  const size_t save = ar.off;
  h16* mb   = (h16*)aalloc(ar, (size_t)Ch * HW * 2);
  h16* Q    = (h16*)aalloc(ar, (size_t)HW * Kw * 2);
  h16* Wn   = (h16*)aalloc(ar, (size_t)Lpad * Kw * 2);
  h16* R    = (h16*)aalloc(ar, (size_t)9 * C * Lpad * 2);
  h16* att  = (h16*)aalloc(ar, (size_t)HW * Lpad * 2);
  float* mB = (float*)aalloc(ar, (size_t)HW * 4);
  float* sB = (float*)aalloc(ar, (size_t)HW * 4);
  h16* ref  = (h16*)aalloc(ar, (size_t)C * HW * 2);
  h16* base = (h16*)aalloc(ar, (size_t)C * HW * 2);
  h16* refm = (h16*)aalloc(ar, (size_t)Ch * HW * 2);

  // match_base = prelu(conv1x1(x, mb))
  conv(st, x, d.mb_w, d.mb_b, d.mb_a, nullptr, mb, nullptr, C, Ch, H, W, 1, CONV_PRELU);
  {
    const size_t n = (size_t)HW * Kw;
    build_q_k<<<(unsigned)cdivz(n, 256), 256, 0, st>>>(mb, Q, Ch, H, W, Kw);
  }
  {
    const size_t n = (size_t)9 * C * Lpad;
    fill0_k<<<(unsigned)cdivz(n, 256), 256, 0, st>>>(R, n);
  }
  for (int i = 0; i < level; ++i) {
    const h16* r = x;
    if (i > 0) {
      const size_t n = (size_t)C * Hs[i] * Wv[i];
      bilinear_k<<<(unsigned)cdivz(n, 256), 256, 0, st>>>(x, ref, C, H, W, Hs[i], Wv[i], 0);
      r = ref;
    }
    conv(st, r, d.as_w, d.as_b, d.as_a, nullptr, base, nullptr, C, C, Hs[i], Wv[i], 1, CONV_PRELU);
    conv(st, r, d.m_w, d.m_b, d.m_a, nullptr, refm, nullptr, C, Ch, Hs[i], Wv[i], 1, CONV_PRELU);
    build_wn_k<<<Hs[i] * Wv[i], 64, 0, st>>>(refm, Wn, Hs[i], Wv[i], Kw, loff[i]);
    {
      const size_t n = (size_t)9 * C * Hs[i] * Wv[i];
      build_r_k<<<(unsigned)cdivz(n, 256), 256, 0, st>>>(base, R, C, Hs[i], Wv[i], Lpad, loff[i]);
    }
  }
  // scores = Q (im2col of match_base) x Wn^T
  {
    dim3 g(cdiv(HW, 32), cdiv(L, 256));
    gemm_nt_wmma_k<<<g, 128, 0, st>>>(Q, Wn, att, HW, L, Kw, Lpad);
  }
  softmax_ms_k<<<cdiv(HW, 256), 256, 0, st>>>(att, mB, sB, HW, L, Lpad);
  {
    const size_t n = (size_t)HW * Lpad;
    softmax_apply_k<<<(unsigned)cdivz(n, 256), 256, 0, st>>>(att, mB, sB, HW, L, Lpad);
  }
  {
    dim3 g(cdiv(HW, 32), cdiv(C, 128));
    pa_aggregate_k<<<g, 64, 0, st>>>(att, R, x, out, C, H, W, Lpad);
  }
  ar.off = save;
}

// down: maxpool -> conv3 relu -> resblock -> PA -> resblock
static void down_stage(hipStream_t st, Arena& ar, const h16* xin, h16* xout,
                       int Cin, int Cout, int H, int W, int level, const DownW& d) {
  const int OH = H / 2, OW = W / 2, OHW = OH * OW;
  const size_t save = ar.off;
  h16* pool = (h16*)aalloc(ar, (size_t)Cin * OHW * 2);
  h16* db   = (h16*)aalloc(ar, (size_t)Cout * OHW * 2);
  h16* tb   = (h16*)aalloc(ar, (size_t)Cout * OHW * 2);
  h16* d2   = (h16*)aalloc(ar, (size_t)Cout * OHW * 2);
  h16* pa   = (h16*)aalloc(ar, (size_t)Cout * OHW * 2);
  {
    const size_t n = (size_t)Cin * OHW;
    maxpool2_k<<<(unsigned)cdivz(n, 256), 256, 0, st>>>(xin, pool, Cin, H, W);
  }
  conv(st, pool, d.bb_w, d.bb_b, nullptr, nullptr, db, nullptr, Cin, Cout, OH, OW, 3, CONV_RELU);
  conv(st, db, d.r1c1_w, d.r1c1_b, nullptr, nullptr, tb, nullptr, Cout, Cout, OH, OW, 3, CONV_RELU);
  conv(st, tb, d.r1c2_w, d.r1c2_b, nullptr, db, d2, nullptr, Cout, Cout, OH, OW, 3, CONV_NONE);
  pyramid_attn(st, ar, d2, pa, Cout, OH, OW, level, d);
  conv(st, pa, d.r2c1_w, d.r2c1_b, nullptr, nullptr, tb, nullptr, Cout, Cout, OH, OW, 3, CONV_RELU);
  conv(st, tb, d.r2c2_w, d.r2c2_b, nullptr, pa, xout, nullptr, Cout, Cout, OH, OW, 3, CONV_NONE);
  ar.off = save;
}

// up: bilinear(align=True) small -> cat([skip, up]) -> conv3 relu (+opt resid)
static void up_stage(hipStream_t st, Arena& ar, const h16* small, const h16* skip,
                     h16* out, int Cs, int Ck, int Cout, int SH, int SW, int OH,
                     int OW, const float* b, h16* w, const h16* resid) {
  const size_t save = ar.off;
  const size_t ohw = (size_t)OH * OW;
  h16* cat = (h16*)aalloc(ar, (size_t)(Ck + Cs) * ohw * 2);
  copy16_k<<<(unsigned)cdivz((size_t)Ck * ohw, 256), 256, 0, st>>>(skip, cat, (size_t)Ck * ohw);
  bilinear_k<<<(unsigned)cdivz((size_t)Cs * ohw, 256), 256, 0, st>>>(
      small, cat + (size_t)Ck * ohw, Cs, SH, SW, OH, OW, 1);
  conv(st, cat, w, b, nullptr, resid, out, nullptr, Ck + Cs, Cout, OH, OW, 3, CONV_RELU);
  ar.off = save;
}

extern "C" void kernel_launch(void* const* d_in, const int* in_sizes, int n_in,
                              void* d_out, int out_size, void* d_ws, size_t ws_size,
                              hipStream_t stream) {
  (void)in_sizes; (void)n_in; (void)out_size;
  Arena ws{(char*)d_ws, 0, ws_size};
  auto F = [&](int i) { return (const float*)d_in[i]; };

  // --- pack all weights (fp32 -> fp16, K padded to 64) ---------------------
  // jax pytree flatten order (alphabetical dict keys):
  // down1 @0, down2 @19, down3 @38, head @57, tail @62, up1 @66, up2 @68,
  // up3 @70, x @72
  DownW dw1 = pack_down(stream, ws, d_in, 0, 64, 128);
  DownW dw2 = pack_down(stream, ws, d_in, 19, 128, 256);
  DownW dw3 = pack_down(stream, ws, d_in, 38, 256, 384);
  const float* head_a = F(57);
  const float* head_c0_b = F(58); h16* head_c0_w = pack_weight(stream, ws, F(59), 64, 4 * 9);
  const float* head_c1_b = F(60); h16* head_c1_w = pack_weight(stream, ws, F(61), 64, 64 * 9);
  const float* tail_c0_b = F(62); h16* tail_c0_w = pack_weight(stream, ws, F(63), 3, 64 * 9);
  const float* tail_c1_b = F(64); h16* tail_c1_w = pack_weight(stream, ws, F(65), 3, 3);
  const float* up1_b = F(66); h16* up1_w = pack_weight(stream, ws, F(67), 64, 192 * 9);
  const float* up2_b = F(68); h16* up2_w = pack_weight(stream, ws, F(69), 128, 384 * 9);
  const float* up3_b = F(70); h16* up3_w = pack_weight(stream, ws, F(71), 256, 640 * 9);
  const float* x_in = F(72);  // [1,4,128,128]

  // --- persistent activations ---------------------------------------------
  h16* x16 = (h16*)aalloc(ws, (size_t)4 * 16384 * 2);
  h16* hb  = (h16*)aalloc(ws, (size_t)64 * 16384 * 2);
  h16* x1  = (h16*)aalloc(ws, (size_t)64 * 16384 * 2);
  h16* x2  = (h16*)aalloc(ws, (size_t)128 * 4096 * 2);
  h16* x3  = (h16*)aalloc(ws, (size_t)256 * 1024 * 2);
  h16* x4  = (h16*)aalloc(ws, (size_t)384 * 256 * 2);
  h16* o3  = (h16*)aalloc(ws, (size_t)256 * 1024 * 2);
  h16* o2  = (h16*)aalloc(ws, (size_t)128 * 4096 * 2);
  h16* o1  = (h16*)aalloc(ws, (size_t)64 * 16384 * 2);
  h16* t256 = (h16*)aalloc(ws, (size_t)64 * 65536 * 2);
  h16* t3  = (h16*)aalloc(ws, (size_t)3 * 65536 * 2);

  // --- head ----------------------------------------------------------------
  cast16_k<<<(unsigned)cdivz((size_t)4 * 16384, 256), 256, 0, stream>>>(x_in, x16, (size_t)4 * 16384);
  conv(stream, x16, head_c0_w, head_c0_b, head_a, nullptr, hb, nullptr, 4, 64, 128, 128, 3, CONV_PRELU);
  conv(stream, hb, head_c1_w, head_c1_b, nullptr, nullptr, x1, nullptr, 64, 64, 128, 128, 3, CONV_NONE);

  // --- encoder -------------------------------------------------------------
  down_stage(stream, ws, x1, x2, 64, 128, 128, 128, 4, dw1);
  down_stage(stream, ws, x2, x3, 128, 256, 64, 64, 3, dw2);
  down_stage(stream, ws, x3, x4, 256, 384, 32, 32, 2, dw3);

  // --- decoder -------------------------------------------------------------
  up_stage(stream, ws, x4, x3, o3, 384, 256, 256, 16, 16, 32, 32, up3_b, up3_w, nullptr);
  up_stage(stream, ws, o3, x2, o2, 256, 128, 128, 32, 32, 64, 64, up2_b, up2_w, nullptr);
  up_stage(stream, ws, o2, x1, o1, 128, 64, 64, 64, 64, 128, 128, up1_b, up1_w, x1);

  // --- tail ----------------------------------------------------------------
  bilinear_k<<<(unsigned)cdivz((size_t)64 * 65536, 256), 256, 0, stream>>>(
      o1, t256, 64, 128, 128, 256, 256, 1);
  conv(stream, t256, tail_c0_w, tail_c0_b, nullptr, nullptr, t3, nullptr, 64, 3, 256, 256, 3, CONV_RELU);
  conv(stream, t3, tail_c1_w, tail_c1_b, nullptr, nullptr, nullptr, (float*)d_out, 3, 3, 256, 256, 1, CONV_NONE);
}
